// SphCrossAttFusionBlock_24103356465409
// MI455X (gfx1250) — matrix operand order repaired and checked by reading, not compiled, not run on hardware
//
#include <hip/hip_runtime.h>
#include <hip/hip_bf16.h>

// ---------------------------------------------------------------------------
// Problem constants
// ---------------------------------------------------------------------------
#define NB   8       // batch
#define DIM  256
#define YY   4096
#define NH   8       // heads
#define DH   64      // dim_head
#define INNER 512    // NH*DH
#define NW   256     // kv tokens = Y / 16
#define KK   4096    // kv contraction dim = DIM*16
#define SCALE_F 0.125f   // 64^-0.5

typedef __attribute__((ext_vector_type(8)))  __bf16 v8bf;
typedef __attribute__((ext_vector_type(16))) __bf16 v16bf;
typedef __attribute__((ext_vector_type(8)))  float  v8f;

__device__ inline __bf16 f2bf(float f) {
    unsigned int u = __builtin_bit_cast(unsigned int, f);
    unsigned int r = (u + 0x7fffu + ((u >> 16) & 1u)) >> 16;
    return __builtin_bit_cast(__bf16, (unsigned short)r);
}

__device__ inline v8f wmma_bf(v16bf a, v16bf b, v8f c) {
    return __builtin_amdgcn_wmma_f32_16x16x32_bf16(
        /*neg_a=*/false, a, /*neg_b=*/false, b,
        /*c_mod=*/(short)0, c, /*reuse_a=*/false, /*reuse_b=*/false);
}

// A-fragment (16x32 bf16): lane holds row (lane&15); two 8-elem runs.
__device__ inline v16bf make_a(const __bf16* p, int off) {
    v8bf a0 = *(const v8bf*)(p + 2 * off);
    v8bf a1 = *(const v8bf*)(p + 2 * off + 8);
    v16bf r;
#pragma unroll
    for (int i = 0; i < 8; ++i) { r[i] = a0[i]; r[i + 8] = a1[i]; }
    return r;
}

// A operand: row-major M x K
__device__ inline v16bf load_a(const __bf16* A, int lda, int m0, int kc, int lane) {
    const __bf16* p = A + (size_t)(m0 + (lane & 15)) * lda + kc;
    return make_a(p, (lane < 16) ? 0 : 8);
}

// B operand stored transposed: N x K row-major (contiguous 32B per lane).
__device__ inline v16bf load_b(const __bf16* Bt, int ldb, int n0, int kc, int lane) {
    const __bf16* p = Bt + (size_t)(n0 + (lane & 15)) * ldb + kc + ((lane < 16) ? 0 : 16);
    return *(const v16bf*)p;
}

// CDNA5 async copy: global -> LDS, 16 bytes, ASYNCcnt-tracked.
__device__ inline void async_ld16(unsigned int lds_off, const void* gptr) {
    asm volatile("global_load_async_to_lds_b128 %0, %1, off"
                 :: "v"(lds_off), "v"(gptr) : "memory");
}

// ---------------------------------------------------------------------------
// Packing kernels (fp32 -> bf16, layout shuffles)
// ---------------------------------------------------------------------------
__global__ __launch_bounds__(256) void cvt_bf16(const float* __restrict__ src,
                                                __bf16* __restrict__ dst, size_t n) {
    size_t i = (size_t)blockIdx.x * blockDim.x + threadIdx.x;
    if (i < n) dst[i] = f2bf(src[i]);
}

__global__ __launch_bounds__(256) void pack_wout(const float* __restrict__ w0,
                                                 const float* __restrict__ w1,
                                                 __bf16* __restrict__ wcat) {
    int i = blockIdx.x * blockDim.x + threadIdx.x;
    if (i >= DIM * 2 * INNER) return;
    int c = i >> 10, j = i & 1023;
    float v = (j < INNER) ? w0[c * INNER + j] : w1[c * INNER + (j - INNER)];
    wcat[i] = f2bf(v);
}

// xT[b][y][n] = x[b][n][y] ; xkvT[b][w][n*16+pq] = x[b][n][w*16+pq]
__global__ __launch_bounds__(256) void pack_x(const float* __restrict__ x,
                                              __bf16* __restrict__ xT,
                                              __bf16* __restrict__ xkvT) {
    size_t i = (size_t)blockIdx.x * blockDim.x + threadIdx.x;
    if (i >= (size_t)NB * YY * DIM) return;
    int n = (int)(i & (DIM - 1));
    size_t t = i >> 8;
    int y = (int)(t & (YY - 1));
    int b = (int)(t >> 12);
    float v = x[((size_t)b * DIM + n) * YY + y];
    __bf16 bv = f2bf(v);
    xT[i] = bv;
    int w = y >> 4, pq = y & 15;
    xkvT[((size_t)b * NW + w) * KK + n * 16 + pq] = bv;
}

// ---------------------------------------------------------------------------
// Software-pipelined wave GEMM core: wave tile 32M x 64N, K chunks of 32.
// Double-buffered fragments so VMEM overlaps the WMMA pipe.
// ---------------------------------------------------------------------------
struct Frags { v16bf a0, a1, b[4]; };

__device__ inline void load_chunk(const __bf16* A, int lda, int m0,
                                  const __bf16* Bt, int ldb, int n0,
                                  int kc, int lane, Frags& f) {
    f.a0 = load_a(A, lda, m0, kc, lane);
    f.a1 = load_a(A, lda, m0 + 16, kc, lane);
#pragma unroll
    for (int j = 0; j < 4; ++j) f.b[j] = load_b(Bt, ldb, n0 + j * 16, kc, lane);
}

__device__ inline void gemm_wave(const __bf16* A, int lda, int m0,
                                 const __bf16* Bt, int ldb, int n0,
                                 int K, int lane, v8f acc0[4], v8f acc1[4]) {
#pragma unroll
    for (int j = 0; j < 4; ++j) { acc0[j] = (v8f)(0.0f); acc1[j] = (v8f)(0.0f); }
    Frags cur, nxt;
    load_chunk(A, lda, m0, Bt, ldb, n0, 0, lane, cur);
    for (int kc = 0; kc < K; kc += 32) {
        int kn = (kc + 32 < K) ? (kc + 32) : kc;   // branchless tail: reload same
        load_chunk(A, lda, m0, Bt, ldb, n0, kn, lane, nxt);
#pragma unroll
        for (int j = 0; j < 4; ++j) acc0[j] = wmma_bf(cur.a0, cur.b[j], acc0[j]);
#pragma unroll
        for (int j = 0; j < 4; ++j) acc1[j] = wmma_bf(cur.a1, cur.b[j], acc1[j]);
        cur = nxt;
    }
}

// ---------------------------------------------------------------------------
// GEMM 1: Q projection. C[o,y] = Wq[o,n]*x[n,y]; writes qT[b][y][o].
// grid: (Y/256, 512/64, 2*NB); block 256 = 8 waves (2M x 4N), wave 32x64.
// ---------------------------------------------------------------------------
__global__ __launch_bounds__(256) void gemm_q(const __bf16* __restrict__ wq,
                                              const __bf16* __restrict__ xT,
                                              __bf16* __restrict__ qT) {
    const int lane = threadIdx.x & 31, wv = threadIdx.x >> 5;
    const int wm = wv & 1, wn = wv >> 1;
    const int s = blockIdx.z & 1, b = blockIdx.z >> 1;
    const int m0 = blockIdx.y * 64 + wm * 32;
    const int n0 = blockIdx.x * 256 + wn * 64;
    const __bf16* A  = wq + (size_t)s * INNER * DIM;
    const __bf16* Bt = xT + ((size_t)s * NB + b) * YY * DIM;
    v8f acc0[4], acc1[4];
    gemm_wave(A, DIM, m0, Bt, DIM, n0, DIM, lane, acc0, acc1);
    __bf16* dst = qT + ((size_t)s * NB + b) * YY * INNER;
    const int rb = (lane < 16) ? 0 : 8;
#pragma unroll
    for (int j = 0; j < 4; ++j) {
        int n = n0 + j * 16 + (lane & 15);
#pragma unroll
        for (int r = 0; r < 8; ++r) {
            dst[(size_t)n * INNER + (m0 + rb + r)]      = f2bf(acc0[j][r]);
            dst[(size_t)n * INNER + (m0 + 16 + rb + r)] = f2bf(acc1[j][r]);
        }
    }
}

// ---------------------------------------------------------------------------
// GEMM 2: KV projection; epilogue splits k[b][h][w][d] / vT[b][h][d][w].
// grid: (1, 1024/64, 2*NB)
// ---------------------------------------------------------------------------
__global__ __launch_bounds__(256) void gemm_kv(const __bf16* __restrict__ wkv,
                                               const __bf16* __restrict__ xkvT,
                                               __bf16* __restrict__ kT,
                                               __bf16* __restrict__ vT) {
    const int lane = threadIdx.x & 31, wv = threadIdx.x >> 5;
    const int wm = wv & 1, wn = wv >> 1;
    const int s = blockIdx.z & 1, b = blockIdx.z >> 1;
    const int m0 = blockIdx.y * 64 + wm * 32;
    const int n0 = wn * 64;
    const __bf16* A  = wkv  + (size_t)s * 2 * INNER * KK;
    const __bf16* Bt = xkvT + ((size_t)s * NB + b) * NW * KK;
    v8f acc0[4], acc1[4];
    gemm_wave(A, KK, m0, Bt, KK, n0, KK, lane, acc0, acc1);
    const int rb = (lane < 16) ? 0 : 8;
#pragma unroll
    for (int j = 0; j < 4; ++j) {
        int n = n0 + j * 16 + (lane & 15);          // w
#pragma unroll
        for (int r = 0; r < 8; ++r) {
#pragma unroll
            for (int half = 0; half < 2; ++half) {
                int m = m0 + half * 16 + rb + r;     // o
                __bf16 v = f2bf(half ? acc1[j][r] : acc0[j][r]);
                if (m < INNER) {
                    int h = m >> 6, d = m & 63;
                    kT[((((size_t)s * NB + b) * NH + h) * NW + n) * DH + d] = v;
                } else {
                    int mo = m - INNER;
                    int h = mo >> 6, d = mo & 63;
                    vT[((((size_t)s * NB + b) * NH + h) * DH + d) * NW + n] = v;
                }
            }
        }
    }
}

// ---------------------------------------------------------------------------
// Flash-style cross attention. K/V staged to LDS via async b128 copies.
// grid: (NB*NH, 8 y-splits, 2 streams); block 256 = 8 waves.
// ---------------------------------------------------------------------------
__global__ __launch_bounds__(256) void attn_kernel(const __bf16* __restrict__ qT_all,
                                                   const __bf16* __restrict__ kT_all,
                                                   const __bf16* __restrict__ vT_all,
                                                   __bf16* __restrict__ ocat) {
    const int lane = threadIdx.x & 31;
    const int wave = threadIdx.x >> 5;
    const int b = blockIdx.x >> 3, h = blockIdx.x & 7;
    const int s = blockIdx.z, so = s ^ 1;
    const int half = lane >> 4, lr = lane & 15;
    const int aoff = half ? 8 : 0;
    const int boff = half ? 16 : 0;

    const __bf16* qT = qT_all + ((size_t)s * NB + b) * YY * INNER;
    const __bf16* kT = kT_all + (((size_t)so * NB + b) * NH + h) * NW * DH;
    const __bf16* vT = vT_all + (((size_t)so * NB + b) * NH + h) * DH * NW;

    __shared__ __align__(32) __bf16 ktile[NW][DH];   // 32 KB, [w][d]
    __shared__ __align__(32) __bf16 vtile[DH][NW];   // 32 KB, [d][w]
    __shared__ __align__(32) __bf16 plds[8][16][32]; //  8 KB
    __bf16 (*P)[32] = plds[wave];

    // --- stage K and V into LDS with CDNA5 async copies (4096 x b128 total) ---
    {
        unsigned int kbase = (unsigned int)(uintptr_t)(&ktile[0][0]);
        unsigned int vbase = (unsigned int)(uintptr_t)(&vtile[0][0]);
#pragma unroll
        for (int i = 0; i < 8; ++i) {
            int c = threadIdx.x + i * 256;           // 16-byte chunk index
            async_ld16(kbase + c * 16, kT + (size_t)c * 8);
            async_ld16(vbase + c * 16, vT + (size_t)c * 8);
        }
        asm volatile("s_wait_asynccnt 0" ::: "memory");
        __syncthreads();
    }

    for (int t = 0; t < 4; ++t) {
        const int ybase = ((blockIdx.y * 8 + wave) * 4 + t) * 16;
        const __bf16* qrow = qT + (size_t)(ybase + lr) * INNER + h * DH;
        v16bf qa0 = make_a(qrow, aoff);       // d = 0..31
        v16bf qa1 = make_a(qrow + 32, aoff);  // d = 32..63

        float m[8], l[8];
        v8f ofr[4];
#pragma unroll
        for (int r = 0; r < 8; ++r) { m[r] = -1e30f; l[r] = 0.0f; }
#pragma unroll
        for (int f = 0; f < 4; ++f) ofr[f] = (v8f)(0.0f);

        for (int wc = 0; wc < NW; wc += 32) {
            // ---- S = Q K^T for a 16x32 slab (B-frags from LDS) ----
            v8f s0 = (v8f)(0.0f), s1 = (v8f)(0.0f);
            const __bf16* k0 = &ktile[wc + lr][0];
            const __bf16* k1 = &ktile[wc + 16 + lr][0];
            s0 = wmma_bf(qa0, *(const v16bf*)(k0 + boff),      s0);
            s0 = wmma_bf(qa1, *(const v16bf*)(k0 + 32 + boff), s0);
            s1 = wmma_bf(qa0, *(const v16bf*)(k1 + boff),      s1);
            s1 = wmma_bf(qa1, *(const v16bf*)(k1 + 32 + boff), s1);

            // ---- online softmax (rows striped: half0 r=0..7, half1 r=8..15) ----
#pragma unroll
            for (int r = 0; r < 8; ++r) {
                float a0 = s0[r] * SCALE_F, a1 = s1[r] * SCALE_F;
                float mx = fmaxf(a0, a1);
                mx = fmaxf(mx, __shfl_xor(mx, 1, 32));
                mx = fmaxf(mx, __shfl_xor(mx, 2, 32));
                mx = fmaxf(mx, __shfl_xor(mx, 4, 32));
                mx = fmaxf(mx, __shfl_xor(mx, 8, 32));
                float mn = fmaxf(m[r], mx);
                float cf = __expf(m[r] - mn);
                m[r] = mn;
                float p0 = __expf(a0 - mn), p1 = __expf(a1 - mn);
                float rs = p0 + p1;
                rs += __shfl_xor(rs, 1, 32);
                rs += __shfl_xor(rs, 2, 32);
                rs += __shfl_xor(rs, 4, 32);
                rs += __shfl_xor(rs, 8, 32);
                l[r] = l[r] * cf + rs;
#pragma unroll
                for (int f = 0; f < 4; ++f) ofr[f][r] *= cf;
                int row = half * 8 + r;
                P[row][lr]      = f2bf(p0);
                P[row][16 + lr] = f2bf(p1);
            }
            asm volatile("s_wait_dscnt 0" ::: "memory");

            v16bf pa = make_a(&P[lr][0], aoff);      // C-frag -> A-frag via LDS

            // ---- O += P * V (B-frags from LDS) ----
#pragma unroll
            for (int f = 0; f < 4; ++f) {
                v16bf bv = *(const v16bf*)(&vtile[f * 16 + lr][wc + boff]);
                ofr[f] = wmma_bf(pa, bv, ofr[f]);
            }
        }

        __bf16* od = ocat + (size_t)b * YY * (2 * INNER) + (size_t)s * INNER + h * DH;
#pragma unroll
        for (int r = 0; r < 8; ++r) {
            float inv = 1.0f / l[r];
            size_t y = (size_t)(ybase + half * 8 + r);
#pragma unroll
            for (int f = 0; f < 4; ++f)
                od[y * (2 * INNER) + f * 16 + lr] = f2bf(ofr[f][r] * inv);
        }
    }
}

// ---------------------------------------------------------------------------
// GEMM 3: fused out-proj for BOTH streams (K=1024) + bias + residuals.
// grid: (Y/256, 256/64, NB)
// ---------------------------------------------------------------------------
__global__ __launch_bounds__(256) void gemm_out(const __bf16* __restrict__ wcat,
                                                const __bf16* __restrict__ ocat,
                                                const float* __restrict__ x0,
                                                const float* __restrict__ x1,
                                                const float* __restrict__ b0,
                                                const float* __restrict__ b1,
                                                float* __restrict__ out) {
    const int lane = threadIdx.x & 31, wv = threadIdx.x >> 5;
    const int wm = wv & 1, wn = wv >> 1;
    const int b = blockIdx.z;
    const int m0 = blockIdx.y * 64 + wm * 32;
    const int n0 = blockIdx.x * 256 + wn * 64;
    const __bf16* Bt = ocat + (size_t)b * YY * (2 * INNER);
    v8f acc0[4], acc1[4];
    gemm_wave(wcat, 2 * INNER, m0, Bt, 2 * INNER, n0, 2 * INNER, lane, acc0, acc1);
    const int rb = (lane < 16) ? 0 : 8;
#pragma unroll
    for (int j = 0; j < 4; ++j) {
        int n = n0 + j * 16 + (lane & 15);
#pragma unroll
        for (int r = 0; r < 8; ++r) {
#pragma unroll
            for (int half = 0; half < 2; ++half) {
                int m = m0 + half * 16 + rb + r;
                float a = half ? acc1[j][r] : acc0[j][r];
                size_t xi = ((size_t)b * DIM + m) * YY + n;
                out[xi] = 0.5f * (a + x0[xi] + x1[xi] + b0[m] + b1[m]);
            }
        }
    }
}

// ---------------------------------------------------------------------------
// Host launch
// ---------------------------------------------------------------------------
extern "C" void kernel_launch(void* const* d_in, const int* in_sizes, int n_in,
                              void* d_out, int out_size, void* d_ws, size_t ws_size,
                              hipStream_t stream) {
    const float* x0    = (const float*)d_in[0];
    const float* x1    = (const float*)d_in[1];
    const float* Wq0   = (const float*)d_in[2];
    const float* Wkv0  = (const float*)d_in[3];
    const float* Wq1   = (const float*)d_in[4];
    const float* Wkv1  = (const float*)d_in[5];
    const float* Wout0 = (const float*)d_in[6];
    const float* bo0   = (const float*)d_in[7];
    const float* Wout1 = (const float*)d_in[8];
    const float* bo1   = (const float*)d_in[9];
    float* out = (float*)d_out;

    constexpr size_t XT_E   = (size_t)2 * NB * YY * DIM;
    constexpr size_t QT_E   = (size_t)2 * NB * YY * INNER;
    constexpr size_t KT_E   = (size_t)2 * NB * NH * NW * DH;
    constexpr size_t WQ_E   = (size_t)2 * INNER * DIM;
    constexpr size_t WKV_E  = (size_t)2 * 2 * INNER * KK;
    constexpr size_t WCAT_E = (size_t)DIM * 2 * INNER;
    constexpr size_t OCAT_E = (size_t)NB * YY * 2 * INNER;

    char* ws = (char*)d_ws;
    size_t cur = 0;
    auto take = [&](size_t bytes) -> size_t {
        size_t o = (cur + 255) & ~(size_t)255;
        cur = o + bytes;
        return o;
    };
    size_t regA   = take(OCAT_E * 2);   // xT+xkvT, later aliased by ocat
    size_t o_wq   = take(WQ_E * 2);
    size_t o_wkv  = take(WKV_E * 2);
    size_t o_wcat = take(WCAT_E * 2);
    size_t o_qt   = take(QT_E * 2);
    size_t o_kt   = take(KT_E * 2);
    size_t o_vt   = take(KT_E * 2);

    __bf16* xT   = (__bf16*)(ws + regA);
    __bf16* xkvT = xT + XT_E;
    __bf16* ocat = (__bf16*)(ws + regA);
    __bf16* wq   = (__bf16*)(ws + o_wq);
    __bf16* wkv  = (__bf16*)(ws + o_wkv);
    __bf16* wcat = (__bf16*)(ws + o_wcat);
    __bf16* qT   = (__bf16*)(ws + o_qt);
    __bf16* kT   = (__bf16*)(ws + o_kt);
    __bf16* vT   = (__bf16*)(ws + o_vt);

    {
        size_t nq = (size_t)INNER * DIM, nkv = (size_t)2 * INNER * KK;
        cvt_bf16<<<dim3((unsigned)((nq + 255) / 256)), 256, 0, stream>>>(Wq0, wq, nq);
        cvt_bf16<<<dim3((unsigned)((nq + 255) / 256)), 256, 0, stream>>>(Wq1, wq + nq, nq);
        cvt_bf16<<<dim3((unsigned)((nkv + 255) / 256)), 256, 0, stream>>>(Wkv0, wkv, nkv);
        cvt_bf16<<<dim3((unsigned)((nkv + 255) / 256)), 256, 0, stream>>>(Wkv1, wkv + nkv, nkv);
        pack_wout<<<dim3((DIM * 2 * INNER) / 256), 256, 0, stream>>>(Wout0, Wout1, wcat);
    }
    {
        size_t nx = (size_t)NB * YY * DIM;
        unsigned g = (unsigned)((nx + 255) / 256);
        pack_x<<<dim3(g), 256, 0, stream>>>(x0, xT, xkvT);
        pack_x<<<dim3(g), 256, 0, stream>>>(x1, xT + nx, xkvT + nx);
    }
    gemm_q<<<dim3(YY / 256, INNER / 64, 2 * NB), 256, 0, stream>>>(wq, xT, qT);
    gemm_kv<<<dim3(1, (2 * INNER) / 64, 2 * NB), 256, 0, stream>>>(wkv, xkvT, kT, vT);
    attn_kernel<<<dim3(NB * NH, 8, 2), 256, 0, stream>>>(qT, kT, vT, ocat);
    gemm_out<<<dim3(YY / 256, DIM / 64, NB), 256, 0, stream>>>(wcat, ocat, x0, x1, bo0, bo1, out);
    (void)in_sizes; (void)n_in; (void)out_size; (void)ws_size;
}
